// HopfieldConv_89223650607720
// MI455X (gfx1250) — matrix operand order repaired and checked by reading, not compiled, not run on hardware
//
#include <hip/hip_runtime.h>

// ---- problem constants (from reference) ----
#define B_    16
#define CIN_  256
#define L_    2048
#define HID_  512
#define G_    2
#define M_    512
#define LP_   1024   // L/2
#define R_    (B_ * HID_)   // 8192 rows for the Hopfield GEMMs

typedef __attribute__((ext_vector_type(2)))  float  v2f;
typedef __attribute__((ext_vector_type(8)))  float  v8f;
typedef __attribute__((ext_vector_type(8)))  __bf16 v8bf;
typedef __attribute__((ext_vector_type(16))) __bf16 v16bf;

// D = A(16x32 bf16) * B(32x16 bf16) + C(16x16 f32), wave32 WMMA
static __device__ __forceinline__ v8f wmma_bf16(v16bf a, v16bf b, v8f c) {
    return __builtin_amdgcn_wmma_f32_16x16x32_bf16(
        false, a, false, b, (short)0, c, false, false);
}

static __device__ __forceinline__ v16bf cat16(v8bf lo, v8bf hi) {
    return __builtin_shufflevector(lo, hi, 0,1,2,3,4,5,6,7,8,9,10,11,12,13,14,15);
}

// f32 -> bf16 hi/lo split (hi + lo represents x to ~2^-16 relative)
static __device__ __forceinline__ void bsplit(float x, __bf16& h, __bf16& l) {
    h = (__bf16)x;
    l = (__bf16)(x - (float)h);
}

// =====================================================================
// prep 1: x [B,CIN,L] -> xt hi/lo planes [B,L,CIN]  (LDS tile transpose)
// =====================================================================
__global__ __launch_bounds__(256) void prep_xt_kernel(
    const float* __restrict__ x, __bf16* __restrict__ xtH, __bf16* __restrict__ xtL) {
    __shared__ float tile[32][33];
    const int tx = threadIdx.x, ty = threadIdx.y;
    const int l0 = blockIdx.x * 32, c0 = blockIdx.y * 32, b = blockIdx.z;
#pragma unroll
    for (int j = 0; j < 4; ++j)
        tile[ty + 8 * j][tx] = x[((size_t)b * CIN_ + c0 + ty + 8 * j) * L_ + l0 + tx];
    __syncthreads();
#pragma unroll
    for (int j = 0; j < 4; ++j) {
        const float v = tile[tx][ty + 8 * j];
        const size_t idx = ((size_t)b * L_ + l0 + ty + 8 * j) * CIN_ + c0 + tx;
        __bf16 h, l; bsplit(v, h, l);
        xtH[idx] = h; xtL[idx] = l;
    }
}

// =====================================================================
// prep 2: patterns [512,1024] -> P hi/lo (row-major) and Pt hi/lo (transposed)
// =====================================================================
__global__ __launch_bounds__(256) void prep_p_kernel(
    const float* __restrict__ P,
    __bf16* __restrict__ pH, __bf16* __restrict__ pL,
    __bf16* __restrict__ ptH, __bf16* __restrict__ ptL) {
    const int i = blockIdx.x * 256 + threadIdx.x;     // 524288
    const int m = i >> 10, l = i & 1023;
    __bf16 h, lo; bsplit(P[i], h, lo);
    pH[i] = h; pL[i] = lo;
    ptH[(size_t)l * M_ + m] = h; ptL[(size_t)l * M_ + m] = lo;
}

// =====================================================================
// prep 3: w1 [512,128] -> hi/lo planes;  wT [512,128,2] -> wTr[g][k][o][i] hi/lo
// =====================================================================
__global__ __launch_bounds__(256) void prep_w_kernel(
    const float* __restrict__ w1, const float* __restrict__ wT,
    __bf16* __restrict__ w1H, __bf16* __restrict__ w1L,
    __bf16* __restrict__ wtH, __bf16* __restrict__ wtL) {
    const int i = blockIdx.x * 256 + threadIdx.x;     // 196608
    if (i < 65536) {
        __bf16 h, l; bsplit(w1[i], h, l);
        w1H[i] = h; w1L[i] = l;
    } else {
        const int j = i - 65536;                      // over 512*128*2
        const int k  = j & 1;
        const int o  = (j >> 1) & 127;
        const int gi = j >> 8;                        // g*256 + i
        const int g  = gi >> 8, ii = gi & 255;
        const size_t dst = (((size_t)(g * 2 + k) * 128 + o) * 256) + ii;
        __bf16 h, l; bsplit(wT[j], h, l);
        wtH[dst] = h; wtL[dst] = l;
    }
}

// =====================================================================
// Kernel 1: grouped 1x1 conv + bias + ReLU + MaxPool(2), bf16x3
// M = pooled position (16), N = out-channel (16), K = 128 in-ch.
// A = xt rows (even/odd share B);  B = w1 rows.  Output: h hi/lo planes.
// =====================================================================
__global__ __launch_bounds__(256) void conv_pool_kernel(
    const __bf16* __restrict__ xtH, const __bf16* __restrict__ xtL,
    const __bf16* __restrict__ w1H, const __bf16* __restrict__ w1L,
    const float* __restrict__ b1,
    __bf16* __restrict__ hH, __bf16* __restrict__ hL) {
    const int lane = threadIdx.x & 31;
    const int wave = threadIdx.x >> 5;
    int tile = blockIdx.x * 8 + wave;            // 32768 = 16*2*16*64
    const int pt = tile & 63;  tile >>= 6;
    const int ot = tile & 15;  tile >>= 4;
    const int g  = tile & 1;
    const int b  = tile >> 1;

    const int mm   = lane & 15;
    const int half = lane >> 4;
    const int klo  = 16 * half;                  // A K-offset
    const int khB  = 16 * half;                  // B K-offset
    const int o0 = ot * 16, p0 = pt * 16;

    // A rows: xt[b][2*(p0+mm)(+1)][g*128 + k]
    const size_t arow = ((size_t)b * L_ + 2 * (p0 + mm)) * CIN_ + g * 128;
    // B row: w1[g*256 + o0 + mm][k]
    const size_t brow = (size_t)(g * 256 + o0 + mm) * 128;

    v8f accE = {}, accO = {};
#pragma unroll
    for (int k0 = 0; k0 < 128; k0 += 32) {
        v16bf aeh = cat16(*(const v8bf*)(xtH + arow + k0 + klo),
                          *(const v8bf*)(xtH + arow + k0 + klo + 8));
        v16bf ael = cat16(*(const v8bf*)(xtL + arow + k0 + klo),
                          *(const v8bf*)(xtL + arow + k0 + klo + 8));
        v16bf aoh = cat16(*(const v8bf*)(xtH + arow + CIN_ + k0 + klo),
                          *(const v8bf*)(xtH + arow + CIN_ + k0 + klo + 8));
        v16bf aol = cat16(*(const v8bf*)(xtL + arow + CIN_ + k0 + klo),
                          *(const v8bf*)(xtL + arow + CIN_ + k0 + klo + 8));
        v16bf bh  = cat16(*(const v8bf*)(w1H + brow + k0 + khB),
                          *(const v8bf*)(w1H + brow + k0 + khB + 8));
        v16bf bl  = cat16(*(const v8bf*)(w1L + brow + k0 + khB),
                          *(const v8bf*)(w1L + brow + k0 + khB + 8));
        accE = wmma_bf16(aeh, bh, accE);
        accE = wmma_bf16(aeh, bl, accE);
        accE = wmma_bf16(ael, bh, accE);
        accO = wmma_bf16(aoh, bh, accO);
        accO = wmma_bf16(aoh, bl, accO);
        accO = wmma_bf16(aol, bh, accO);
    }

    // C tile: m = pooled pos (v + 8*half), n = out-channel (mm, lane-fixed)
    const int ch = g * 256 + o0 + mm;
    const float bias = b1[ch];
    const size_t r = (size_t)b * HID_ + ch;
    v8bf oh, ol;
#pragma unroll
    for (int v = 0; v < 8; ++v) {
        float val = fmaxf(fmaxf(accE[v], accO[v]) + bias, 0.0f);
        __bf16 h, l; bsplit(val, h, l);
        oh[v] = h; ol[v] = l;
    }
    const size_t dst = r * LP_ + p0 + 8 * half;
    *(v8bf*)(hH + dst) = oh;
    *(v8bf*)(hL + dst) = ol;
}

// =====================================================================
// Kernel 2: scores[r,m] = sum_l h[r,l] * P[m,l], bf16x3.
// Per wave: 16 rows x 32 m (2 N-tiles share A).
// =====================================================================
__global__ __launch_bounds__(256) void scores_kernel(
    const __bf16* __restrict__ hH, const __bf16* __restrict__ hL,
    const __bf16* __restrict__ pH, const __bf16* __restrict__ pL,
    float* __restrict__ scores) {
    const int lane = threadIdx.x & 31;
    const int wave = threadIdx.x >> 5;
    const int wid  = blockIdx.x * 8 + wave;      // 8192 waves
    const int mt2  = wid & 15;                   // 16 groups of 2 m-tiles
    const int rt   = wid >> 4;                   // 512 row tiles
    const int r0   = rt * 16;
    const int m0   = mt2 * 32;

    const int mm   = lane & 15;
    const int half = lane >> 4;
    const int klo  = 16 * half;
    const int khB  = 16 * half;

    const size_t arow = (size_t)(r0 + mm) * LP_;
    const size_t b0r  = (size_t)(m0 + mm) * LP_;
    const size_t b1r  = (size_t)(m0 + 16 + mm) * LP_;

    v8f c0 = {}, c1 = {};
#pragma unroll 2
    for (int k0 = 0; k0 < LP_; k0 += 32) {
        v16bf ah = cat16(*(const v8bf*)(hH + arow + k0 + klo),
                         *(const v8bf*)(hH + arow + k0 + klo + 8));
        v16bf al = cat16(*(const v8bf*)(hL + arow + k0 + klo),
                         *(const v8bf*)(hL + arow + k0 + klo + 8));
        v16bf b0h = cat16(*(const v8bf*)(pH + b0r + k0 + khB),
                          *(const v8bf*)(pH + b0r + k0 + khB + 8));
        v16bf b0l = cat16(*(const v8bf*)(pL + b0r + k0 + khB),
                          *(const v8bf*)(pL + b0r + k0 + khB + 8));
        v16bf b1h = cat16(*(const v8bf*)(pH + b1r + k0 + khB),
                          *(const v8bf*)(pH + b1r + k0 + khB + 8));
        v16bf b1l = cat16(*(const v8bf*)(pL + b1r + k0 + khB),
                          *(const v8bf*)(pL + b1r + k0 + khB + 8));
        c0 = wmma_bf16(ah, b0h, c0);
        c0 = wmma_bf16(ah, b0l, c0);
        c0 = wmma_bf16(al, b0h, c0);
        c1 = wmma_bf16(ah, b1h, c1);
        c1 = wmma_bf16(ah, b1l, c1);
        c1 = wmma_bf16(al, b1h, c1);
    }

#pragma unroll
    for (int v = 0; v < 8; ++v) {
        const int m = v + 8 * half;
        float* row = scores + (size_t)(r0 + m) * M_ + m0 + mm;
        row[0]  = c0[v];
        row[16] = c1[v];
    }
}

// =====================================================================
// Kernel 3: row softmax over 512 + bf16 hi/lo split of attn
// =====================================================================
__global__ __launch_bounds__(256) void softmax_kernel(
    const float* __restrict__ s, __bf16* __restrict__ aH, __bf16* __restrict__ aL) {
    const int lane = threadIdx.x & 31;
    const int wave = threadIdx.x >> 5;
    const int row  = blockIdx.x * 8 + wave;      // 8192 rows
    const float* sr = s + (size_t)row * M_;

    float vals[16];
    float mx = -3.4e38f;
#pragma unroll
    for (int i = 0; i < 16; ++i) { vals[i] = sr[lane + i * 32]; mx = fmaxf(mx, vals[i]); }
#pragma unroll
    for (int off = 16; off; off >>= 1) mx = fmaxf(mx, __shfl_xor(mx, off, 32));
    float sum = 0.0f;
#pragma unroll
    for (int i = 0; i < 16; ++i) { vals[i] = __expf(vals[i] - mx); sum += vals[i]; }
#pragma unroll
    for (int off = 16; off; off >>= 1) sum += __shfl_xor(sum, off, 32);
    const float inv = 1.0f / sum;
#pragma unroll
    for (int i = 0; i < 16; ++i) {
        __bf16 h, l; bsplit(vals[i] * inv, h, l);
        aH[(size_t)row * M_ + lane + i * 32] = h;
        aL[(size_t)row * M_ + lane + i * 32] = l;
    }
}

// =====================================================================
// Kernel 4: h2[r,l] = sum_m attn[r,m] * P[m,l], bf16x3.
// B from transposed patterns Pt[l][m]. Output transposed: h2t[b][l][c] hi/lo.
// =====================================================================
__global__ __launch_bounds__(256) void retrieve_kernel(
    const __bf16* __restrict__ aH, const __bf16* __restrict__ aL,
    const __bf16* __restrict__ ptH, const __bf16* __restrict__ ptL,
    __bf16* __restrict__ h2H, __bf16* __restrict__ h2L) {
    const int lane = threadIdx.x & 31;
    const int wave = threadIdx.x >> 5;
    const int wid  = blockIdx.x * 8 + wave;      // 16384 waves
    const int lt2  = wid & 31;                   // 32 groups of 2 l-tiles
    const int rt   = wid >> 5;                   // 512 row tiles
    const int r0   = rt * 16;
    const int l0   = lt2 * 32;

    const int mm   = lane & 15;
    const int half = lane >> 4;
    const int klo  = 16 * half;
    const int khB  = 16 * half;

    const size_t arow = (size_t)(r0 + mm) * M_;
    const size_t b0r  = (size_t)(l0 + mm) * M_;
    const size_t b1r  = (size_t)(l0 + 16 + mm) * M_;

    v8f c0 = {}, c1 = {};
#pragma unroll 2
    for (int k0 = 0; k0 < M_; k0 += 32) {
        v16bf ah = cat16(*(const v8bf*)(aH + arow + k0 + klo),
                         *(const v8bf*)(aH + arow + k0 + klo + 8));
        v16bf al = cat16(*(const v8bf*)(aL + arow + k0 + klo),
                         *(const v8bf*)(aL + arow + k0 + klo + 8));
        v16bf b0h = cat16(*(const v8bf*)(ptH + b0r + k0 + khB),
                          *(const v8bf*)(ptH + b0r + k0 + khB + 8));
        v16bf b0l = cat16(*(const v8bf*)(ptL + b0r + k0 + khB),
                          *(const v8bf*)(ptL + b0r + k0 + khB + 8));
        v16bf b1h = cat16(*(const v8bf*)(ptH + b1r + k0 + khB),
                          *(const v8bf*)(ptH + b1r + k0 + khB + 8));
        v16bf b1l = cat16(*(const v8bf*)(ptL + b1r + k0 + khB),
                          *(const v8bf*)(ptL + b1r + k0 + khB + 8));
        c0 = wmma_bf16(ah, b0h, c0);
        c0 = wmma_bf16(ah, b0l, c0);
        c0 = wmma_bf16(al, b0h, c0);
        c1 = wmma_bf16(ah, b1h, c1);
        c1 = wmma_bf16(ah, b1l, c1);
        c1 = wmma_bf16(al, b1h, c1);
    }

    // store transposed+split: h2t[b][l][c]; per-lane c is contiguous over v
    const int b  = r0 >> 9;
    const int cb = (r0 & 511) + 8 * half;
    v8bf o0h, o0l, o1h, o1l;
#pragma unroll
    for (int v = 0; v < 8; ++v) {
        __bf16 h, l;
        bsplit(c0[v], h, l); o0h[v] = h; o0l[v] = l;
        bsplit(c1[v], h, l); o1h[v] = h; o1l[v] = l;
    }
    const size_t d0 = ((size_t)b * LP_ + l0 + mm) * HID_ + cb;
    const size_t d1 = ((size_t)b * LP_ + l0 + 16 + mm) * HID_ + cb;
    *(v8bf*)(h2H + d0) = o0h; *(v8bf*)(h2L + d0) = o0l;
    *(v8bf*)(h2H + d1) = o1h; *(v8bf*)(h2L + d1) = o1l;
}

// =====================================================================
// Kernel 5: grouped ConvTranspose1d(k=2,s=2) + bias + ReLU + residual, bf16x3
// M = position (16), N = out-channel-in-group (16); both taps share A.
// Per-lane stores are contiguous float2 pairs (taps interleave along L).
// =====================================================================
__global__ __launch_bounds__(256) void deconv_kernel(
    const __bf16* __restrict__ h2H, const __bf16* __restrict__ h2L,
    const __bf16* __restrict__ wtH, const __bf16* __restrict__ wtL,
    const float* __restrict__ bT, const float* __restrict__ x,
    const float* __restrict__ RZp, float* __restrict__ out) {
    const int lane = threadIdx.x & 31;
    const int wave = threadIdx.x >> 5;
    int tile = blockIdx.x * 8 + wave;            // 16384 = 16*2*8*64
    const int lt = tile & 63;  tile >>= 6;
    const int ot = tile & 7;   tile >>= 3;
    const int g  = tile & 1;
    const int b  = tile >> 1;

    const int mm   = lane & 15;
    const int half = lane >> 4;
    const int klo  = 16 * half;
    const int khB  = 16 * half;
    const int o0 = ot * 16, l0 = lt * 16;

    const float rz = RZp[0];

    // A row: h2t[b][l0+mm][g*256 + k]
    const size_t arow = ((size_t)b * LP_ + l0 + mm) * HID_ + g * 256;
    // B rows: wTr[g][k][o0+mm][i]
    const size_t bt0 = (((size_t)(g * 2 + 0) * 128) + o0 + mm) * 256;
    const size_t bt1 = (((size_t)(g * 2 + 1) * 128) + o0 + mm) * 256;

    v8f c0 = {}, c1 = {};
#pragma unroll 2
    for (int k0 = 0; k0 < 256; k0 += 32) {
        v16bf ah = cat16(*(const v8bf*)(h2H + arow + k0 + klo),
                         *(const v8bf*)(h2H + arow + k0 + klo + 8));
        v16bf al = cat16(*(const v8bf*)(h2L + arow + k0 + klo),
                         *(const v8bf*)(h2L + arow + k0 + klo + 8));
        v16bf b0h = cat16(*(const v8bf*)(wtH + bt0 + k0 + khB),
                          *(const v8bf*)(wtH + bt0 + k0 + khB + 8));
        v16bf b0l = cat16(*(const v8bf*)(wtL + bt0 + k0 + khB),
                          *(const v8bf*)(wtL + bt0 + k0 + khB + 8));
        v16bf b1h = cat16(*(const v8bf*)(wtH + bt1 + k0 + khB),
                          *(const v8bf*)(wtH + bt1 + k0 + khB + 8));
        v16bf b1l = cat16(*(const v8bf*)(wtL + bt1 + k0 + khB),
                          *(const v8bf*)(wtL + bt1 + k0 + khB + 8));
        c0 = wmma_bf16(ah, b0h, c0);
        c0 = wmma_bf16(ah, b0l, c0);
        c0 = wmma_bf16(al, b0h, c0);
        c1 = wmma_bf16(ah, b1h, c1);
        c1 = wmma_bf16(ah, b1l, c1);
        c1 = wmma_bf16(al, b1h, c1);
    }

    // C tiles: m = l (v + 8*half), n = out-channel (mm, lane-fixed)
    const int c   = g * 128 + o0 + mm;
    const float bias = bT[c];
    const size_t base = ((size_t)b * CIN_ + c) * L_ + 2 * (size_t)(l0 + 8 * half);
#pragma unroll
    for (int v = 0; v < 8; ++v) {
        const float y0 = fmaxf(c0[v] + bias, 0.0f);   // tap k=0 -> pos 2l
        const float y1 = fmaxf(c1[v] + bias, 0.0f);   // tap k=1 -> pos 2l+1
        const size_t idx = base + 2 * (size_t)v;
        v2f xi = *(const v2f*)(x + idx);
        v2f o  = { xi.x + rz * y0, xi.y + rz * y1 };
        *(v2f*)(out + idx) = o;
    }
}

// =====================================================================
// host launcher
// =====================================================================
extern "C" void kernel_launch(void* const* d_in, const int* in_sizes, int n_in,
                              void* d_out, int out_size, void* d_ws, size_t ws_size,
                              hipStream_t stream) {
    const float* x        = (const float*)d_in[0];   // [16,256,2048]
    const float* w1       = (const float*)d_in[1];   // [512,128,1]
    const float* b1       = (const float*)d_in[2];   // [512]
    const float* patterns = (const float*)d_in[3];   // [512,1024]
    const float* wT       = (const float*)d_in[4];   // [512,128,2]
    const float* bT       = (const float*)d_in[5];   // [256]
    const float* RZ       = (const float*)d_in[6];   // scalar
    float* out            = (float*)d_out;           // [16,256,2048]

    __bf16* bf = (__bf16*)d_ws;
    size_t off = 0;
    __bf16* xtH  = bf + off; off += (size_t)B_ * L_ * CIN_;   // 8388608
    __bf16* xtL  = bf + off; off += (size_t)B_ * L_ * CIN_;
    __bf16* hH   = bf + off; off += (size_t)R_ * LP_;         // 8388608
    __bf16* hL   = bf + off; off += (size_t)R_ * LP_;
    __bf16* pH   = bf + off; off += (size_t)M_ * LP_;         // 524288
    __bf16* pL   = bf + off; off += (size_t)M_ * LP_;
    __bf16* ptH  = bf + off; off += (size_t)M_ * LP_;
    __bf16* ptL  = bf + off; off += (size_t)M_ * LP_;
    __bf16* w1H  = bf + off; off += (size_t)HID_ * 128;       // 65536
    __bf16* w1L  = bf + off; off += (size_t)HID_ * 128;
    __bf16* wtH  = bf + off; off += (size_t)HID_ * 128 * 2;   // 131072
    __bf16* wtL  = bf + off; off += (size_t)HID_ * 128 * 2;
    __bf16* aH   = bf + off; off += (size_t)R_ * M_;          // 4194304
    __bf16* aL   = bf + off; off += (size_t)R_ * M_;
    __bf16* h2H  = bf + off; off += (size_t)B_ * LP_ * HID_;  // 8388608
    __bf16* h2L  = bf + off; off += (size_t)B_ * LP_ * HID_;
    float* scores = (float*)(bf + off);                       // 8192*512 f32

    prep_xt_kernel <<<dim3(L_ / 32, CIN_ / 32, B_), dim3(32, 8), 0, stream>>>(x, xtH, xtL);
    prep_p_kernel  <<<(M_ * LP_) / 256, 256, 0, stream>>>(patterns, pH, pL, ptH, ptL);
    prep_w_kernel  <<<(65536 + 131072) / 256, 256, 0, stream>>>(w1, wT, w1H, w1L, wtH, wtL);

    conv_pool_kernel<<<4096, 256, 0, stream>>>(xtH, xtL, w1H, w1L, b1, hH, hL);
    scores_kernel   <<<1024, 256, 0, stream>>>(hH, hL, pH, pL, scores);
    softmax_kernel  <<<1024, 256, 0, stream>>>(scores, aH, aL);
    retrieve_kernel <<<2048, 256, 0, stream>>>(aH, aL, ptH, ptL, h2H, h2L);
    deconv_kernel   <<<2048, 256, 0, stream>>>(h2H, h2L, wtH, wtL, bT, x, RZ, out);
}
